// MoELayer_50843822850159
// MI455X (gfx1250) — matrix operand order repaired and checked by reading, not compiled, not run on hardware
//
#include <hip/hip_runtime.h>
#include <hip/hip_bf16.h>

#define DI __device__ __forceinline__

typedef __attribute__((ext_vector_type(16))) __bf16 v16bf;
typedef __attribute__((ext_vector_type(8)))  float  v8f;
typedef int v4i_ __attribute__((vector_size(16)));          // matches builtin param type
typedef __attribute__((address_space(1))) v4i_* g4p_t;      // global v4i*
typedef __attribute__((address_space(3))) v4i_* l4p_t;      // lds v4i*

#if defined(__has_builtin)
#if __has_builtin(__builtin_amdgcn_global_load_async_to_lds_b128) && \
    __has_builtin(__builtin_amdgcn_s_wait_asynccnt)
#define USE_ASYNC_LDS 1
#endif
#endif
#ifndef USE_ASYNC_LDS
#define USE_ASYNC_LDS 0
#endif

namespace {
constexpr int Bc = 2, Sc = 2048, Hc = 1024, Ec = 8, Kk = 2, Fc = 4096;
constexpr int Tc = Bc * Sc;      // 4096 tokens
constexpr int RBLK = Tc / 256;   // 16 router blocks
constexpr float AUX_COEF = 0.001f, Z_COEF = 0.001f;

// d_out layout (floats): [out T*H][aux 1][route_probs T*K][route_indices T*K]
constexpr size_t AUX_OFF  = (size_t)Tc * Hc;
constexpr size_t PROB_OFF = AUX_OFF + 1;
constexpr size_t IDX_OFF  = PROB_OFF + (size_t)Tc * Kk;

// workspace byte offsets
constexpr size_t WS_COUNTS  = 0;                                  // E ints
constexpr size_t WS_BASES   = 64;                                 // E ints
constexpr size_t WS_REDF    = 128;                                // RBLK*E floats
constexpr size_t WS_REDP    = 128 + 512;                          // RBLK*E floats
constexpr size_t WS_REDZ    = 1152;                               // RBLK floats
constexpr size_t WS_TOK     = 1280;                               // E*T ints
constexpr size_t WS_SLOT    = WS_TOK  + (size_t)Ec * Tc * 4;
constexpr size_t WS_GATE    = WS_SLOT + (size_t)Ec * Tc * 4;
constexpr size_t WS_HMID    = ((WS_GATE + (size_t)Ec * Tc * 4 + 255) / 256) * 256; // T*K x F bf16
constexpr size_t WS_CONTRIB = WS_HMID + (size_t)Tc * Kk * Fc * 2;                  // T*K x H f32

constexpr int LDA = 40;   // LDS row stride in bf16 elements (80B) - conflict padding
} // namespace

DI __bf16 f2bf(float f) { return (__bf16)f; }

DI v16bf ldfrag(const __bf16* p0, const __bf16* p1) {
  v16bf r;
  uint4* q = reinterpret_cast<uint4*>(&r);
  q[0] = *reinterpret_cast<const uint4*>(p0);
  q[1] = *reinterpret_cast<const uint4*>(p1);
  return r;
}

DI float gelu_exact(float v) { return 0.5f * v * (1.0f + erff(v * 0.70710678118654752f)); }

// ---------------------------------------------------------------- init
__global__ void moe_init(int* counts) {
  if (threadIdx.x < Ec) counts[threadIdx.x] = 0;
}

// ---------------------------------------------------------------- router
DI void block_reduce_write(float* red, float v, float* dst, int tid) {
  red[tid] = v;
  __syncthreads();
  for (int s = 128; s > 0; s >>= 1) {
    if (tid < s) red[tid] += red[tid + s];
    __syncthreads();
  }
  if (tid == 0) *dst = red[0];
  __syncthreads();
}

__global__ __launch_bounds__(256) void moe_router(
    const float* __restrict__ x, const float* __restrict__ wr,
    float* __restrict__ out, int* __restrict__ counts,
    int* __restrict__ tok, int* __restrict__ slt, float* __restrict__ gate,
    float* __restrict__ redf, float* __restrict__ redp, float* __restrict__ redz) {
  __shared__ float wr_s[Hc * Ec];
  __shared__ float red[256];
  const int tid = threadIdx.x;
  for (int i = tid; i < Hc * Ec; i += 256) wr_s[i] = wr[i];
  __syncthreads();

  const int t = blockIdx.x * 256 + tid;
  float acc[Ec] = {};
  const float4* xr = reinterpret_cast<const float4*>(x + (size_t)t * Hc);
  for (int hq = 0; hq < Hc / 4; ++hq) {
    float4 xv = xr[hq];
    float xa[4] = {xv.x, xv.y, xv.z, xv.w};
#pragma unroll
    for (int j = 0; j < 4; ++j) {
      const float4 w0 = *reinterpret_cast<const float4*>(&wr_s[(hq * 4 + j) * Ec]);
      const float4 w1v = *reinterpret_cast<const float4*>(&wr_s[(hq * 4 + j) * Ec + 4]);
      const float xs = xa[j];
      acc[0] += xs * w0.x;  acc[1] += xs * w0.y;  acc[2] += xs * w0.z;  acc[3] += xs * w0.w;
      acc[4] += xs * w1v.x; acc[5] += xs * w1v.y; acc[6] += xs * w1v.z; acc[7] += xs * w1v.w;
    }
  }
  // softmax over E=8
  float m = acc[0];
#pragma unroll
  for (int e = 1; e < Ec; ++e) m = fmaxf(m, acc[e]);
  float p[Ec], s = 0.f;
#pragma unroll
  for (int e = 0; e < Ec; ++e) { p[e] = __expf(acc[e] - m); s += p[e]; }
  const float inv = 1.f / s;
#pragma unroll
  for (int e = 0; e < Ec; ++e) p[e] *= inv;

  // top-2 (first-occurrence tie-break like lax.top_k)
  int i1 = 0; float p1 = p[0];
#pragma unroll
  for (int e = 1; e < Ec; ++e) if (p[e] > p1) { p1 = p[e]; i1 = e; }
  int i2 = -1; float p2 = -1.f;
#pragma unroll
  for (int e = 0; e < Ec; ++e) if (e != i1 && p[e] > p2) { p2 = p[e]; i2 = e; }

  out[PROB_OFF + (size_t)t * 2]     = p1;
  out[PROB_OFF + (size_t)t * 2 + 1] = p2;
  out[IDX_OFF  + (size_t)t * 2]     = (float)i1;
  out[IDX_OFF  + (size_t)t * 2 + 1] = (float)i2;

  int pos = atomicAdd(&counts[i1], 1);
  tok[i1 * Tc + pos] = t; slt[i1 * Tc + pos] = t * 2 + 0; gate[i1 * Tc + pos] = p1;
  pos = atomicAdd(&counts[i2], 1);
  tok[i2 * Tc + pos] = t; slt[i2 * Tc + pos] = t * 2 + 1; gate[i2 * Tc + pos] = p2;

  const float lse = __logf(__expf(p1) + __expf(p2));
  const float zv = lse * lse;

  // deterministic per-block reductions for aux loss
  for (int e = 0; e < Ec; ++e)
    block_reduce_write(red, p[e], &redp[blockIdx.x * Ec + e], tid);
  for (int e = 0; e < Ec; ++e)
    block_reduce_write(red, (e == i1 || e == i2) ? 1.f : 0.f, &redf[blockIdx.x * Ec + e], tid);
  block_reduce_write(red, zv, &redz[blockIdx.x], tid);
}

// ---------------------------------------------------------------- finalize (bases + aux loss)
__global__ void moe_finalize(const int* __restrict__ counts, int* __restrict__ bases,
                             const float* __restrict__ redf, const float* __restrict__ redp,
                             const float* __restrict__ redz, float* __restrict__ out) {
  if (threadIdx.x == 0 && blockIdx.x == 0) {
    int b = 0;
    for (int e = 0; e < Ec; ++e) { bases[e] = b; b += counts[e]; }
    float lb = 0.f;
    for (int e = 0; e < Ec; ++e) {
      float fs = 0.f, ps = 0.f;
      for (int k = 0; k < RBLK; ++k) { fs += redf[k * Ec + e]; ps += redp[k * Ec + e]; }
      lb += (fs / (float)Tc) * (ps / (float)Tc);
    }
    float zs = 0.f;
    for (int k = 0; k < RBLK; ++k) zs += redz[k];
    out[AUX_OFF] = (float)Ec * lb * AUX_COEF + (zs / (float)Tc) * Z_COEF;
  }
}

// ---------------------------------------------------------------- FFN1: hmid = gelu(Xe @ W1e + b1e)
__global__ __launch_bounds__(256) void moe_ffn1(
    const float* __restrict__ x, const float* __restrict__ w1,
    const float* __restrict__ b1, const int* __restrict__ counts,
    const int* __restrict__ bases, const int* __restrict__ tok,
    unsigned short* __restrict__ hmid_raw) {
  const int e = blockIdx.z;
  const int cnt = counts[e];
  const int m0 = blockIdx.y * 128;
  if (m0 >= cnt) return;
  const int bse = bases[e];
  const int n0 = blockIdx.x * 128;
  __bf16* hmid = reinterpret_cast<__bf16*>(hmid_raw);

  __shared__ __bf16 Asl[128 * LDA];
  __shared__ __bf16 Bsl[128 * LDA];

  const int tid = threadIdx.x;
  const int lane = tid & 31;
  const int wid = tid >> 5;
  const int wm = (wid & 3) * 32;    // wave M offset in tile
  const int wn = (wid >> 2) * 64;   // wave N offset in tile
  const int lh = lane & 15;
  const bool hiL = lane >= 16;

  // A staging: thread -> (row, half), 16 f32 each (gather by token id, cvt bf16)
  const int arow = tid >> 1, ahalf = tid & 1;
  int srow = m0 + arow; if (srow >= cnt) srow = cnt - 1;
  const int tokid = tok[e * Tc + srow];
  const float* xsrc = x + (size_t)tokid * Hc + ahalf * 16;
  // B staging: thread -> (k-pair, n-chunk of 8); software transpose into LDS [n][k]
  const int kp = tid >> 4, nc = tid & 15;
  const float* wsrc = w1 + ((size_t)e * Hc + 2 * kp) * Fc + n0 + nc * 8;

  v8f acc[2][4] = {};

  for (int kb = 0; kb < Hc; kb += 32) {
    float4 ax0 = *reinterpret_cast<const float4*>(xsrc + kb);
    float4 ax1 = *reinterpret_cast<const float4*>(xsrc + kb + 4);
    float4 ax2 = *reinterpret_cast<const float4*>(xsrc + kb + 8);
    float4 ax3 = *reinterpret_cast<const float4*>(xsrc + kb + 12);
    const float* wk = wsrc + (size_t)kb * Fc;
    float4 br00 = *reinterpret_cast<const float4*>(wk);
    float4 br01 = *reinterpret_cast<const float4*>(wk + 4);
    float4 br10 = *reinterpret_cast<const float4*>(wk + Fc);
    float4 br11 = *reinterpret_cast<const float4*>(wk + Fc + 4);
    if (kb + 32 < Hc) {
      __builtin_prefetch(wk + (size_t)32 * Fc, 0, 3);   // keep in near caches (L2-resident weights)
      __builtin_prefetch(xsrc + kb + 32, 0, 3);
    }
    __syncthreads();
    {
      alignas(16) __bf16 tmp[16];
      float av[16] = {ax0.x, ax0.y, ax0.z, ax0.w, ax1.x, ax1.y, ax1.z, ax1.w,
                      ax2.x, ax2.y, ax2.z, ax2.w, ax3.x, ax3.y, ax3.z, ax3.w};
#pragma unroll
      for (int j = 0; j < 16; ++j) tmp[j] = f2bf(av[j]);
      uint4* d = reinterpret_cast<uint4*>(&Asl[arow * LDA + ahalf * 16]);
      d[0] = reinterpret_cast<uint4*>(tmp)[0];
      d[1] = reinterpret_cast<uint4*>(tmp)[1];

      float r0[8] = {br00.x, br00.y, br00.z, br00.w, br01.x, br01.y, br01.z, br01.w};
      float r1[8] = {br10.x, br10.y, br10.z, br10.w, br11.x, br11.y, br11.z, br11.w};
#pragma unroll
      for (int j = 0; j < 8; ++j) {
        const int n = nc * 8 + j;
        union { __bf16 h[2]; unsigned u; } pk;
        pk.h[0] = f2bf(r0[j]); pk.h[1] = f2bf(r1[j]);
        *reinterpret_cast<unsigned*>(&Bsl[n * LDA + 2 * kp]) = pk.u;
      }
    }
    __syncthreads();

    v16bf afr[2], bfr[4];
    const int ka = hiL ? 8 : 0;       // A frag: lanes 0-15 K{0..7,16..23}; 16-31 K{8..15,24..31}
    afr[0] = ldfrag(&Asl[(wm + lh) * LDA + ka],      &Asl[(wm + lh) * LDA + ka + 16]);
    afr[1] = ldfrag(&Asl[(wm + 16 + lh) * LDA + ka], &Asl[(wm + 16 + lh) * LDA + ka + 16]);
    const int kbb = hiL ? 8 : 0;      // B frag: lanes 0-15 K{0..7,16..23}; 16-31 K{8..15,24..31}
#pragma unroll
    for (int ni = 0; ni < 4; ++ni) {
      const int col = wn + ni * 16 + lh;
      bfr[ni] = ldfrag(&Bsl[col * LDA + kbb], &Bsl[col * LDA + kbb + 16]);
    }
#pragma unroll
    for (int mi = 0; mi < 2; ++mi)
#pragma unroll
      for (int ni = 0; ni < 4; ++ni)
        acc[mi][ni] = __builtin_amdgcn_wmma_f32_16x16x32_bf16(
            false, afr[mi], false, bfr[ni], (short)0, acc[mi][ni], false, false);
  }

  // epilogue: +b1, exact gelu, store bf16 to grouped hmid rows
  const int mAdd = hiL ? 8 : 0;
#pragma unroll
  for (int mi = 0; mi < 2; ++mi)
#pragma unroll
    for (int ni = 0; ni < 4; ++ni) {
      const int f = n0 + wn + ni * 16 + lh;
      const float bias = b1[(size_t)e * Fc + f];
#pragma unroll
      for (int r = 0; r < 8; ++r) {
        const int mrow = wm + mi * 16 + mAdd + r;
        if (m0 + mrow < cnt) {
          float v = acc[mi][ni][r] + bias;
          hmid[(size_t)(bse + m0 + mrow) * Fc + f] = f2bf(gelu_exact(v));
        }
      }
    }
}

// ---------------------------------------------------------------- FFN2: contrib = gate * (hmid @ W2e + b2e)
__global__ __launch_bounds__(256) void moe_ffn2(
    const unsigned short* __restrict__ hmid_raw, const float* __restrict__ w2,
    const float* __restrict__ b2, const int* __restrict__ counts,
    const int* __restrict__ bases, const float* __restrict__ gate,
    const int* __restrict__ slt, float* __restrict__ contrib) {
  const int e = blockIdx.z;
  const int cnt = counts[e];
  const int m0 = blockIdx.y * 128;
  if (m0 >= cnt) return;
  const int bse = bases[e];
  const int n0 = blockIdx.x * 128;
  const __bf16* hmid = reinterpret_cast<const __bf16*>(hmid_raw);

  __shared__ __bf16 Asl[128 * LDA];
  __shared__ __bf16 Bsl[128 * LDA];

  const int tid = threadIdx.x;
  const int lane = tid & 31;
  const int wid = tid >> 5;
  const int wm = (wid & 3) * 32;
  const int wn = (wid >> 2) * 64;
  const int lh = lane & 15;
  const bool hiL = lane >= 16;

  const int arow = tid >> 1, ahalf = tid & 1;
  int srow = m0 + arow; if (srow >= cnt) srow = cnt - 1;
  const __bf16* asrc = hmid + (size_t)(bse + srow) * Fc + ahalf * 16;
  const int kp = tid >> 4, nc = tid & 15;
  const float* wsrc = w2 + ((size_t)e * Fc + 2 * kp) * Hc + n0 + nc * 8;

  v8f acc[2][4] = {};

  for (int kb = 0; kb < Fc; kb += 32) {
#if !USE_ASYNC_LDS
    uint4 a0 = *reinterpret_cast<const uint4*>(asrc + kb);
    uint4 a1 = *reinterpret_cast<const uint4*>(asrc + kb + 8);
#endif
    const float* wk = wsrc + (size_t)kb * Hc;
    float4 br00 = *reinterpret_cast<const float4*>(wk);
    float4 br01 = *reinterpret_cast<const float4*>(wk + 4);
    float4 br10 = *reinterpret_cast<const float4*>(wk + Hc);
    float4 br11 = *reinterpret_cast<const float4*>(wk + Hc + 4);
    if (kb + 32 < Fc) {
      __builtin_prefetch(wk + (size_t)32 * Hc, 0, 3);
      __builtin_prefetch(asrc + kb + 32, 0, 3);
    }
    __syncthreads();   // previous iteration's fragment loads are done: LDS reusable
    {
#if USE_ASYNC_LDS
      // A operand is bf16 in memory already: move it with the gfx1250 async
      // global->LDS path (ASYNCcnt), no VGPR round-trip, overlaps with B cvt below.
      __builtin_amdgcn_global_load_async_to_lds_b128(
          (g4p_t)(asrc + kb),
          (l4p_t)&Asl[arow * LDA + ahalf * 16], 0, 0);
      __builtin_amdgcn_global_load_async_to_lds_b128(
          (g4p_t)(asrc + kb + 8),
          (l4p_t)&Asl[arow * LDA + ahalf * 16 + 8], 0, 0);
#else
      uint4* d = reinterpret_cast<uint4*>(&Asl[arow * LDA + ahalf * 16]);
      d[0] = a0; d[1] = a1;
#endif
      float r0[8] = {br00.x, br00.y, br00.z, br00.w, br01.x, br01.y, br01.z, br01.w};
      float r1[8] = {br10.x, br10.y, br10.z, br10.w, br11.x, br11.y, br11.z, br11.w};
#pragma unroll
      for (int j = 0; j < 8; ++j) {
        const int n = nc * 8 + j;
        union { __bf16 h[2]; unsigned u; } pk;
        pk.h[0] = f2bf(r0[j]); pk.h[1] = f2bf(r1[j]);
        *reinterpret_cast<unsigned*>(&Bsl[n * LDA + 2 * kp]) = pk.u;
      }
    }
#if USE_ASYNC_LDS
    __builtin_amdgcn_s_wait_asynccnt(0);
#endif
    __syncthreads();

    v16bf afr[2], bfr[4];
    const int ka = hiL ? 8 : 0;
    afr[0] = ldfrag(&Asl[(wm + lh) * LDA + ka],      &Asl[(wm + lh) * LDA + ka + 16]);
    afr[1] = ldfrag(&Asl[(wm + 16 + lh) * LDA + ka], &Asl[(wm + 16 + lh) * LDA + ka + 16]);
    const int kbb = hiL ? 8 : 0;
#pragma unroll
    for (int ni = 0; ni < 4; ++ni) {
      const int col = wn + ni * 16 + lh;
      bfr[ni] = ldfrag(&Bsl[col * LDA + kbb], &Bsl[col * LDA + kbb + 16]);
    }
#pragma unroll
    for (int mi = 0; mi < 2; ++mi)
#pragma unroll
      for (int ni = 0; ni < 4; ++ni)
        acc[mi][ni] = __builtin_amdgcn_wmma_f32_16x16x32_bf16(
            false, afr[mi], false, bfr[ni], (short)0, acc[mi][ni], false, false);
  }

  const int mAdd = hiL ? 8 : 0;
#pragma unroll
  for (int mi = 0; mi < 2; ++mi)
#pragma unroll
    for (int ni = 0; ni < 4; ++ni) {
      const int h = n0 + wn + ni * 16 + lh;
      const float bias = b2[(size_t)e * Hc + h];
#pragma unroll
      for (int r = 0; r < 8; ++r) {
        const int mrow = wm + mi * 16 + mAdd + r;
        if (m0 + mrow < cnt) {
          const int g = e * Tc + m0 + mrow;
          const float v = (acc[mi][ni][r] + bias) * gate[g];
          contrib[(size_t)slt[g] * Hc + h] = v;   // per-assignment slot: no atomics
        }
      }
    }
}

// ---------------------------------------------------------------- combine: out[t] = contrib[2t] + contrib[2t+1]
__global__ __launch_bounds__(256) void moe_combine(const float* __restrict__ contrib,
                                                   float* __restrict__ out) {
  const size_t i = (size_t)blockIdx.x * 256 + threadIdx.x;  // over T * (H/4)
  const size_t t = i / (Hc / 4), q = i % (Hc / 4);
  const float4* c = reinterpret_cast<const float4*>(contrib);
  const float4 a = c[(t * 2) * (Hc / 4) + q];
  const float4 b = c[(t * 2 + 1) * (Hc / 4) + q];
  float4 r; r.x = a.x + b.x; r.y = a.y + b.y; r.z = a.z + b.z; r.w = a.w + b.w;
  reinterpret_cast<float4*>(out)[t * (Hc / 4) + q] = r;
}

// ---------------------------------------------------------------- launcher
extern "C" void kernel_launch(void* const* d_in, const int* in_sizes, int n_in,
                              void* d_out, int out_size, void* d_ws, size_t ws_size,
                              hipStream_t stream) {
  const float* x  = (const float*)d_in[0];
  const float* wr = (const float*)d_in[1];
  const float* w1 = (const float*)d_in[2];
  const float* b1 = (const float*)d_in[3];
  const float* w2 = (const float*)d_in[4];
  const float* b2 = (const float*)d_in[5];
  float* out = (float*)d_out;
  char* ws = (char*)d_ws;

  int*   counts  = (int*)(ws + WS_COUNTS);
  int*   bases   = (int*)(ws + WS_BASES);
  float* redf    = (float*)(ws + WS_REDF);
  float* redp    = (float*)(ws + WS_REDP);
  float* redz    = (float*)(ws + WS_REDZ);
  int*   tok     = (int*)(ws + WS_TOK);
  int*   slt     = (int*)(ws + WS_SLOT);
  float* gate    = (float*)(ws + WS_GATE);
  unsigned short* hmid = (unsigned short*)(ws + WS_HMID);
  float* contrib = (float*)(ws + WS_CONTRIB);

  moe_init<<<1, 32, 0, stream>>>(counts);
  moe_router<<<RBLK, 256, 0, stream>>>(x, wr, out, counts, tok, slt, gate, redf, redp, redz);
  moe_finalize<<<1, 32, 0, stream>>>(counts, bases, redf, redp, redz, out);
  moe_ffn1<<<dim3(Fc / 128, Tc / 128, Ec), 256, 0, stream>>>(x, w1, b1, counts, bases, tok, hmid);
  moe_ffn2<<<dim3(Hc / 128, Tc / 128, Ec), 256, 0, stream>>>(hmid, w2, b2, counts, bases, gate, slt, contrib);
  moe_combine<<<(Tc * (Hc / 4)) / 256, 256, 0, stream>>>(contrib, out);

  (void)in_sizes; (void)n_in; (void)out_size; (void)ws_size;
}
